// _StandaloneFIRLoop_19490561590061
// MI455X (gfx1250) — matrix operand order repaired and checked by reading, not compiled, not run on hardware
//
#include <hip/hip_runtime.h>

// ---------------------------------------------------------------------------
// MI455X (gfx1250) FIR-via-WMMA implementation.
//
// Stage 1: zi = x @ zeros_init.T  (only cols 0..254 kept, flipped -> history)
// Stage 2: y[b,t] = sum_k b[k] * x_ext[b, t+255-k]  as banded GEMM vs W[288x16]
//
// Precision: bf16 hi/lo split (x = hi + lo), D = Ah*Bh + Ah*Bl + Al*Bh with
// f32 WMMA accumulation -> ~2^-16 relative error, f32-class.
//
// Workspace layout (bytes):
//   xe_hi : 4096*8704*2 = 71,303,168   (x_ext hi plane, stride 8704)
//   xe_lo : 71,303,168                 (x_ext lo plane)
//   z_hi  : 256*8192*2  =  4,194,304
//   z_lo  :  4,194,304
//   w_hi  : 16*288*2    =      9,216   (banded FIR matrix, column-major)
//   w_lo  :      9,216
// total ~151 MB.
// ---------------------------------------------------------------------------

typedef __attribute__((ext_vector_type(16))) __bf16 bf16x16;
typedef __attribute__((ext_vector_type(8)))  float  f32x8;

#define XB 4096
#define XT 8192
#define XN 256
#define XE_STRIDE 8704           // 255 history + 8192 signal + zero tail (mult of 32)
#define WK 288                   // FIR K padded to multiple of 32

static __device__ __forceinline__ float bf2f(__bf16 h) {
  unsigned short s = __builtin_bit_cast(unsigned short, h);
  unsigned u = ((unsigned)s) << 16;
  return __builtin_bit_cast(float, u);
}
static __device__ __forceinline__ __bf16 f2bf(float f) {  // round-to-nearest-even
  unsigned u = __builtin_bit_cast(unsigned, f);
  unsigned r = (u + 0x7FFFu + ((u >> 16) & 1u)) >> 16;
  return __builtin_bit_cast(__bf16, (unsigned short)r);
}

static __device__ __forceinline__ f32x8 wmma_bf16(bf16x16 a, bf16x16 b, f32x8 c) {
  return __builtin_amdgcn_wmma_f32_16x16x32_bf16(false, a, false, b, (short)0, c,
                                                 false, false);
}

// A fragment (16x32 bf16): as called here (same K order as B), lane grp=lane>>4 holds K-chunks
// [grp*16, grp*16+8) in elements 0..7 and [grp*16+8, grp*16+16) in elements 8..15.
static __device__ __forceinline__ bf16x16 load_a_frag(const __bf16* p0,
                                                      const __bf16* p1) {
  bf16x16 v;
  __builtin_memcpy(&v, p0, 16);
  __builtin_memcpy(reinterpret_cast<char*>(&v) + 16, p1, 16);
  return v;
}

// B fragment (32x16 bf16): lane n holds column n&15, K = (n>>4)*16 + 0..15
// (contiguous when the B matrix column is contiguous in memory).
static __device__ __forceinline__ bf16x16 load_b_frag(const __bf16* p) {
  bf16x16 v;
  __builtin_memcpy(&v, __builtin_assume_aligned(p, 32), 32);
  return v;
}

// ---------------------------------------------------------------------------
// P1: split x (f32) into x_ext hi/lo bf16 planes at offset 255; zero the tail.
// Covers i in [255, XE_STRIDE) per row; cols [0,255) are written by zi_gemm.
// ---------------------------------------------------------------------------
__global__ void __launch_bounds__(256) split_x_kernel(const float* __restrict__ x,
                                                      __bf16* __restrict__ xeh,
                                                      __bf16* __restrict__ xel) {
  const long long nper = XE_STRIDE - 255;  // 8449
  long long idx = (long long)blockIdx.x * 256 + threadIdx.x;
  if (idx >= (long long)XB * nper) return;
  int b = (int)(idx / nper);
  int i = (int)(idx % nper) + 255;                      // [255, 8704)
  float f = (i < 255 + XT) ? x[(size_t)b * XT + (i - 255)] : 0.0f;
  __bf16 h = f2bf(f);
  size_t o = (size_t)b * XE_STRIDE + i;
  xeh[o] = h;
  xel[o] = f2bf(f - bf2f(h));
}

// ---------------------------------------------------------------------------
// P2: split zeros_init (f32, [256][8192]) into hi/lo bf16 planes.
// ---------------------------------------------------------------------------
__global__ void __launch_bounds__(256) split_z_kernel(const float* __restrict__ z,
                                                      __bf16* __restrict__ zh,
                                                      __bf16* __restrict__ zl) {
  int idx = blockIdx.x * 256 + threadIdx.x;
  if (idx >= XN * XT) return;
  float f = z[idx];
  __bf16 h = f2bf(f);
  zh[idx] = h;
  zl[idx] = f2bf(f - bf2f(h));
}

// ---------------------------------------------------------------------------
// P3: banded FIR matrix W[u,j] = b[255-(u-j)] for 0 <= u-j <= 255, else 0.
// Stored column-major: w[j*WK + u], split hi/lo.
// ---------------------------------------------------------------------------
__global__ void __launch_bounds__(256) build_w_kernel(const float* __restrict__ coef,
                                                      __bf16* __restrict__ wh,
                                                      __bf16* __restrict__ wl) {
  int idx = blockIdx.x * 256 + threadIdx.x;
  if (idx >= 16 * WK) return;
  int u = idx % WK, j = idx / WK;
  int d = u - j;
  float w = (d >= 0 && d < 256) ? coef[255 - d] : 0.0f;
  __bf16 h = f2bf(w);
  wh[idx] = h;
  wl[idx] = f2bf(w - bf2f(h));
}

// ---------------------------------------------------------------------------
// Stage 1: zi = x @ zeros_init.T via bf16-split WMMA.
// One block = one 16-row batch tile; 8 waves; wave w computes column tiles
// {w, w+8} (A-fragment reused for 2 tiles). Result column c is written flipped
// into xe[:, 254-c] as hi/lo (c==255 dropped, matching the reference).
// ---------------------------------------------------------------------------
__global__ void __launch_bounds__(256) zi_gemm_kernel(const __bf16* __restrict__ zh,
                                                      const __bf16* __restrict__ zl,
                                                      __bf16* xeh, __bf16* xel) {
  const int lane = threadIdx.x & 31;
  const int wave = threadIdx.x >> 5;      // 0..7
  const int rt   = blockIdx.x;            // 0..255 batch row tile
  const int m    = lane & 15;
  const int grp  = lane >> 4;

  const __bf16* arow_h = xeh + (size_t)(rt * 16 + m) * XE_STRIDE + 255;  // x row
  const __bf16* arow_l = xel + (size_t)(rt * 16 + m) * XE_STRIDE + 255;
  const __bf16* b0h = zh + (size_t)((wave    ) * 16 + m) * XT;   // B col = z row
  const __bf16* b0l = zl + (size_t)((wave    ) * 16 + m) * XT;
  const __bf16* b1h = zh + (size_t)((wave + 8) * 16 + m) * XT;
  const __bf16* b1l = zl + (size_t)((wave + 8) * 16 + m) * XT;

  f32x8 acc0a = {}, acc0b = {}, acc1a = {}, acc1b = {};

#pragma unroll 2
  for (int k0 = 0; k0 < XT; k0 += 32) {
    const int ka = k0 + grp * 16;
    const int kb = k0 + grp * 16;
    __builtin_prefetch(arow_h + ka + 512, 0, 0);   // global_prefetch_b8
    bf16x16 ah = load_a_frag(arow_h + ka, arow_h + ka + 8);
    bf16x16 al = load_a_frag(arow_l + ka, arow_l + ka + 8);
    bf16x16 v0h = load_b_frag(b0h + kb);
    bf16x16 v0l = load_b_frag(b0l + kb);
    bf16x16 v1h = load_b_frag(b1h + kb);
    bf16x16 v1l = load_b_frag(b1l + kb);
    acc0a = wmma_bf16(ah, v0h, acc0a);
    acc0b = wmma_bf16(ah, v0l, acc0b);
    acc0b = wmma_bf16(al, v0h, acc0b);
    acc1a = wmma_bf16(ah, v1h, acc1a);
    acc1b = wmma_bf16(ah, v1l, acc1b);
    acc1b = wmma_bf16(al, v1h, acc1b);
  }

#pragma unroll
  for (int pair = 0; pair < 2; ++pair) {
    int c = (wave + pair * 8) * 16 + m;          // zi column
    if (c > 254) continue;                       // last column shifted out
    f32x8 aa = pair ? acc1a : acc0a;
    f32x8 bb = pair ? acc1b : acc0b;
#pragma unroll
    for (int r = 0; r < 8; ++r) {
      size_t row = (size_t)(rt * 16 + grp * 8 + r);
      float v = aa[r] + bb[r];
      __bf16 h = f2bf(v);
      xeh[row * XE_STRIDE + (254 - c)] = h;
      xel[row * XE_STRIDE + (254 - c)] = f2bf(v - bf2f(h));
    }
  }
}

// ---------------------------------------------------------------------------
// Stage 2: FIR as banded GEMM. One wave per 16x16 output tile:
//   y[b0+m, t0+j] = sum_{u<288} xe[b0+m, t0+u] * W[u,j]
// 9 K-steps of 32, 3 WMMAs each (hi/lo split). All loads 16B/32B aligned.
// ---------------------------------------------------------------------------
__global__ void __launch_bounds__(256) fir_wmma_kernel(const __bf16* __restrict__ xeh,
                                                       const __bf16* __restrict__ xel,
                                                       const __bf16* __restrict__ wh,
                                                       const __bf16* __restrict__ wl,
                                                       float* __restrict__ y) {
  const int lane = threadIdx.x & 31;
  const int wave = threadIdx.x >> 5;           // 0..7
  const int gid  = blockIdx.x * 8 + wave;      // 0..131071
  const int rt   = gid >> 9;                   // batch tile  0..255
  const int t0   = (gid & 511) << 4;           // time offset 0..8176
  const int m    = lane & 15;
  const int grp  = lane >> 4;

  const __bf16* arow_h = xeh + (size_t)(rt * 16 + m) * XE_STRIDE + t0;
  const __bf16* arow_l = xel + (size_t)(rt * 16 + m) * XE_STRIDE + t0;
  const __bf16* wcol_h = wh + m * WK;
  const __bf16* wcol_l = wl + m * WK;

  f32x8 accA = {}, accB = {};

#pragma unroll
  for (int s = 0; s < 9; ++s) {
    const int ka = s * 32 + grp * 16;
    const int kb = s * 32 + grp * 16;
    bf16x16 ah = load_a_frag(arow_h + ka, arow_h + ka + 8);
    bf16x16 al = load_a_frag(arow_l + ka, arow_l + ka + 8);
    bf16x16 bh = load_b_frag(wcol_h + kb);
    bf16x16 bl = load_b_frag(wcol_l + kb);
    accA = wmma_bf16(ah, bh, accA);
    accB = wmma_bf16(ah, bl, accB);
    accB = wmma_bf16(al, bh, accB);
  }

#pragma unroll
  for (int r = 0; r < 8; ++r) {
    y[(size_t)(rt * 16 + grp * 8 + r) * XT + t0 + m] = accA[r] + accB[r];
  }
}

// ---------------------------------------------------------------------------
extern "C" void kernel_launch(void* const* d_in, const int* in_sizes, int n_in,
                              void* d_out, int out_size, void* d_ws, size_t ws_size,
                              hipStream_t stream) {
  (void)in_sizes; (void)n_in; (void)out_size; (void)ws_size;
  const float* x    = (const float*)d_in[0];
  const float* coef = (const float*)d_in[1];
  const float* z    = (const float*)d_in[2];
  float* y = (float*)d_out;

  char* ws = (char*)d_ws;
  const size_t XE_BYTES = (size_t)XB * XE_STRIDE * 2;   // 71,303,168
  const size_t Z_BYTES  = (size_t)XN * XT * 2;          //  4,194,304
  const size_t W_BYTES  = (size_t)16 * WK * 2;          //      9,216
  __bf16* xeh = (__bf16*)(ws);
  __bf16* xel = (__bf16*)(ws + XE_BYTES);
  __bf16* zh  = (__bf16*)(ws + 2 * XE_BYTES);
  __bf16* zl  = (__bf16*)(ws + 2 * XE_BYTES + Z_BYTES);
  __bf16* wh  = (__bf16*)(ws + 2 * XE_BYTES + 2 * Z_BYTES);
  __bf16* wl  = (__bf16*)(ws + 2 * XE_BYTES + 2 * Z_BYTES + W_BYTES);

  {  // P1: x -> x_ext planes
    long long n = (long long)XB * (XE_STRIDE - 255);
    int blocks = (int)((n + 255) / 256);
    split_x_kernel<<<blocks, 256, 0, stream>>>(x, xeh, xel);
  }
  split_z_kernel<<<(XN * XT + 255) / 256, 256, 0, stream>>>(z, zh, zl);
  build_w_kernel<<<(16 * WK + 255) / 256, 256, 0, stream>>>(coef, wh, wl);

  zi_gemm_kernel<<<XB / 16, 256, 0, stream>>>(zh, zl, xeh, xel);

  fir_wmma_kernel<<<(XB / 16) * (XT / 16) / 8, 256, 0, stream>>>(xeh, xel, wh, wl, y);
}
